// CombSub_63118839382098
// MI455X (gfx1250) — compile-verified, matrix operand order checked by reading, not run on hardware
//
#include <hip/hip_runtime.h>
#include <hip/hip_bf16.h>
#include <math.h>

// ---------------- constants ----------------
#define SRATE     44100.0f
#define BATCH     8
#define NFRAMES   800
#define NTOK      (BATCH * NFRAMES)      // 6400 tokens
#define TLEN      (NFRAMES * 512)        // 409600 samples per batch row
#define HEADS     8
#define HDIM      32
#define PI_F      3.14159265358979323846f

typedef __attribute__((ext_vector_type(16))) __bf16 v16bf;
typedef __attribute__((ext_vector_type(8)))  float  v8f;

// ---------------- helpers ----------------
__device__ __forceinline__ unsigned short f2bf(float x) {
    unsigned int u = __builtin_bit_cast(unsigned int, x);
    return (unsigned short)((u + 0x7FFFu + ((u >> 16) & 1u)) >> 16);
}
// CDNA5 16-bit A/B operand K index for VGPR v (0..7), lane half (0/1): pairs (K,K+1)
__device__ __forceinline__ int kb16(int v, int half) {
    return (v < 4) ? (2 * v + 8 * half) : (16 + 2 * (v - 4) + 8 * half);
}

union U16 { v16bf v; unsigned int d[8]; unsigned short u[16]; };

// ---------------- weight packing ----------------
// dense (K,N) row-major f32 -> pair-interleaved bf16 dwords: dst[(k/2)*N+n] = {W[k][n],W[k+1][n]}
__global__ void k_pack_pairs(const float* __restrict__ src, unsigned int* __restrict__ dst,
                             int K, int N) {
    int i = blockIdx.x * 256 + threadIdx.x;
    if (i >= (K >> 1) * N) return;
    int k2 = i / N, n = i - k2 * N;
    unsigned int lo = f2bf(src[(size_t)(2 * k2) * N + n]);
    unsigned int hi = f2bf(src[(size_t)(2 * k2 + 1) * N + n]);
    dst[i] = lo | (hi << 16);
}
// conv weights (O=256,I=256,K=3); effective B row r = k*256+i, col o; pair-interleaved dwords
__global__ void k_pack_conv_pairs(const float* __restrict__ w, unsigned int* __restrict__ dst) {
    int i = blockIdx.x * 256 + threadIdx.x;           // over 384*256
    if (i >= 384 * 256) return;
    int k2 = i >> 8, o = i & 255;
    int r0 = 2 * k2, r1 = r0 + 1;
    int ka = r0 >> 8, ia = r0 & 255;
    int kb = r1 >> 8, ib = r1 & 255;
    unsigned int lo = f2bf(w[(size_t)o * 768 + ia * 3 + ka]);
    unsigned int hi = f2bf(w[(size_t)o * 768 + ib * 3 + kb]);
    dst[i] = lo | (hi << 16);
}

// ---------------- generic bf16 WMMA GEMM (N,K,ACT compile-time) ----------------
// C[M,N] = A[M,K](bf16 row) @ Bp(pair-interleaved) + bias ; ACT: 0 none, 2 tanh-gelu ; +resid
template <int N, int K, int ACT>
__global__ void __launch_bounds__(256)
k_gemm(const unsigned short* __restrict__ A, const unsigned int* __restrict__ Bp,
       const float* __restrict__ bias, const float* __restrict__ resid,
       float* __restrict__ Cf, unsigned short* __restrict__ C16, int M) {
    int wave = (int)(blockIdx.x * blockDim.x + threadIdx.x) >> 5;
    int lane = threadIdx.x & 31;
    constexpr int tilesN = N >> 4;
    int tiles = (M >> 4) * tilesN;
    if (wave >= tiles) return;                 // wave-uniform -> EXEC stays all-ones
    int tm = (wave / tilesN) << 4;
    int tn = (wave - (wave / tilesN) * tilesN) << 4;
    int m = lane & 15, half = lane >> 4;

    v8f acc = {};
    const unsigned short* arow = A + (tm + m) * K;       // A pairs contiguous in K
    const unsigned int*   bcol = Bp + tn + m;            // column base, immediate row offsets
#pragma unroll 4
    for (int k0 = 0; k0 < K; k0 += 32) {
        U16 a, b;
        int krow = (k0 >> 1) * N;
#pragma unroll
        for (int v = 0; v < 8; ++v) {
            int kb = kb16(v, half);
            a.d[v] = *(const unsigned int*)(arow + k0 + kb);
            b.d[v] = bcol[krow + (kb >> 1) * N];
        }
        if (k0 + 32 < K) __builtin_prefetch(arow + k0 + 32, 0, 1);   // global_prefetch_b8
        acc = __builtin_amdgcn_wmma_f32_16x16x32_bf16(false, a.v, false, b.v,
                                                      (short)0, acc, false, false);
    }
#pragma unroll
    for (int r = 0; r < 8; ++r) {
        int mm = tm + r + 8 * half;
        int nn = tn + m;
        float x = acc[r];
        if (bias) x += bias[nn];
        if (ACT == 2) {
            float t = 0.7978845608f * (x + 0.044715f * x * x * x);
            x = 0.5f * x * (1.0f + tanhf(t));
        }
        int idx = mm * N + nn;
        if (resid) x += resid[idx];
        if (C16) C16[idx] = f2bf(x); else Cf[idx] = x;
    }
}

template <int N, int K, int ACT>
static inline void launch_gemm(const unsigned short* A, const unsigned int* Bp,
                               const float* bias, const float* resid,
                               float* Cf, unsigned short* C16, int M, hipStream_t s) {
    int waves = (M / 16) * (N / 16);
    int blocks = (waves * 32 + 255) / 256;
    k_gemm<N, K, ACT><<<blocks, 256, 0, s>>>(A, Bp, bias, resid, Cf, C16, M);
}

// ---------------- im2col (k=3, same pad along frames, per batch row) ----------------
__global__ void k_im2col(const float* __restrict__ src, unsigned short* __restrict__ dst) {
    int t = blockIdx.x, c = threadIdx.x;
    int b = t / NFRAMES, f = t - b * NFRAMES;
#pragma unroll
    for (int k = 0; k < 3; ++k) {
        int ff = f + k - 1;
        float v = (ff >= 0 && ff < NFRAMES) ? src[((size_t)b * NFRAMES + ff) * 256 + c] : 0.0f;
        dst[(size_t)t * 768 + k * 256 + c] = f2bf(v);
    }
}

// ---------------- GroupNorm(groups=4) + LeakyReLU ----------------
__global__ void __launch_bounds__(256)
k_groupnorm_leaky(const float* __restrict__ x, const float* __restrict__ g,
                  const float* __restrict__ bt, float* __restrict__ y) {
    __shared__ float red[256];
    int b = blockIdx.x >> 2, gi = blockIdx.x & 3, tid = threadIdx.x;
    float s = 0.f, s2 = 0.f;
    for (int e = tid; e < 64 * NFRAMES; e += 256) {
        int c = gi * 64 + (e & 63), f = e >> 6;
        float v = x[((size_t)b * NFRAMES + f) * 256 + c];
        s += v; s2 += v * v;
    }
    red[tid] = s; __syncthreads();
    for (int o = 128; o > 0; o >>= 1) { if (tid < o) red[tid] += red[tid + o]; __syncthreads(); }
    float mean = red[0] * (1.0f / (64.0f * NFRAMES)); __syncthreads();
    red[tid] = s2; __syncthreads();
    for (int o = 128; o > 0; o >>= 1) { if (tid < o) red[tid] += red[tid + o]; __syncthreads(); }
    float var = red[0] * (1.0f / (64.0f * NFRAMES)) - mean * mean;
    float inv = rsqrtf(var + 1e-5f);
    for (int e = tid; e < 64 * NFRAMES; e += 256) {
        int c = gi * 64 + (e & 63), f = e >> 6;
        size_t idx = ((size_t)b * NFRAMES + f) * 256 + c;
        float v = (x[idx] - mean) * inv * g[c] + bt[c];
        y[idx] = (v >= 0.f) ? v : 0.01f * v;
    }
}

// ---------------- LayerNorm (D=256) -> bf16 ----------------
__global__ void __launch_bounds__(256)
k_layernorm(const float* __restrict__ x, const float* __restrict__ g,
            const float* __restrict__ b, unsigned short* __restrict__ y) {
    __shared__ float red[256];
    int t = blockIdx.x, d = threadIdx.x;
    float v = x[(size_t)t * 256 + d];
    red[d] = v; __syncthreads();
    for (int o = 128; o > 0; o >>= 1) { if (d < o) red[d] += red[d + o]; __syncthreads(); }
    float mean = red[0] * (1.0f / 256.0f); __syncthreads();
    float c = v - mean;
    red[d] = c * c; __syncthreads();
    for (int o = 128; o > 0; o >>= 1) { if (d < o) red[d] += red[d + o]; __syncthreads(); }
    float inv = rsqrtf(red[0] * (1.0f / 256.0f) + 1e-5f);
    y[(size_t)t * 256 + d] = f2bf(c * inv * g[d] + b[d]);
}

// ---------------- conditioning embeds ----------------
__global__ void k_cond_add(float* __restrict__ xa, const float* __restrict__ f0f,
                           const float* __restrict__ phase, const float* __restrict__ vol,
                           const float* __restrict__ f0w, const float* __restrict__ f0b,
                           const float* __restrict__ phw, const float* __restrict__ phb,
                           const float* __restrict__ vw,  const float* __restrict__ vb) {
    int t = blockIdx.x, d = threadIdx.x;
    float sf = log1pf(f0f[t] * (1.0f / 700.0f));
    float sp = phase[t] * (1.0f / PI_F);
    float sv = vol[t];
    xa[(size_t)t * 256 + d] += sf * f0w[d] + f0b[d] + sp * phw[d] + phb[d] + sv * vw[d] + vb[d];
}

// ---------------- V transpose: qkv16 V section -> Vt[b][d][t] (pairs over tokens) ----------
__global__ void k_vt(const unsigned short* __restrict__ qkv16, unsigned short* __restrict__ vt) {
    int i = blockIdx.x * 256 + threadIdx.x;           // over NTOK*256
    if (i >= NTOK * 256) return;
    int d = i & 255;
    int t = (i >> 8) % NFRAMES;
    int b = i / (256 * NFRAMES);
    vt[((size_t)(b * 256 + d)) * NFRAMES + t] = qkv16[(size_t)(b * NFRAMES + t) * 768 + 512 + d];
}

// ---------------- attention: one wave per (b, h, 16-query block) ----------------
__global__ void __launch_bounds__(32)
k_attn(const unsigned short* __restrict__ qkv16, const unsigned short* __restrict__ vt,
       unsigned short* __restrict__ out16) {
    __shared__ float S[16][NFRAMES];        // 51.2 KB
    const int qb = blockIdx.x % (NFRAMES / 16);
    const int h  = (blockIdx.x / (NFRAMES / 16)) % HEADS;
    const int b  = blockIdx.x / ((NFRAMES / 16) * HEADS);
    const int lane = threadIdx.x;
    const int m = lane & 15, half = lane >> 4;
    const int q0 = qb * 16;

    U16 a;
    const unsigned short* qbase = qkv16 + (size_t)(b * NFRAMES + q0 + m) * 768 + h * HDIM;
#pragma unroll
    for (int v = 0; v < 8; ++v) a.d[v] = *(const unsigned int*)(qbase + kb16(v, half));

    const float scale = 0.17677669529663687f;  // 1/sqrt(32)
    for (int kc = 0; kc < NFRAMES; kc += 16) {
        U16 bm;
        const unsigned short* kbase = qkv16 + (size_t)(b * NFRAMES + kc + m) * 768 + 256 + h * HDIM;
#pragma unroll
        for (int v = 0; v < 8; ++v)
            bm.d[v] = *(const unsigned int*)(kbase + kb16(v, half));
        v8f c = {};
        c = __builtin_amdgcn_wmma_f32_16x16x32_bf16(false, a.v, false, bm.v, (short)0, c, false, false);
#pragma unroll
        for (int r = 0; r < 8; ++r) S[r + 8 * half][kc + m] = c[r] * scale;
    }
    // softmax over 800 per row (wave-cooperative)
    for (int r = 0; r < 16; ++r) {
        float mx = -1e30f;
        for (int c = lane; c < NFRAMES; c += 32) mx = fmaxf(mx, S[r][c]);
        for (int o = 16; o > 0; o >>= 1) mx = fmaxf(mx, __shfl_xor(mx, o));
        float sum = 0.f;
        for (int c = lane; c < NFRAMES; c += 32) { float e = __expf(S[r][c] - mx); S[r][c] = e; sum += e; }
        for (int o = 16; o > 0; o >>= 1) sum += __shfl_xor(sum, o);
        float inv = 1.0f / sum;
        for (int c = lane; c < NFRAMES; c += 32) S[r][c] *= inv;
    }
    // O = P @ V  (V from transposed layout: token pairs contiguous)
    const unsigned short* vtb = vt + ((size_t)b * 256 + h * HDIM) * NFRAMES;
    v8f o0 = {}, o1 = {};
    for (int kc = 0; kc < NFRAMES; kc += 32) {
        U16 p, v0, v1;
#pragma unroll
        for (int v = 0; v < 8; ++v) {
            int kk = kb16(v, half);
            p.u[2 * v]     = f2bf(S[m][kc + kk]);
            p.u[2 * v + 1] = f2bf(S[m][kc + kk + 1]);
            v0.d[v] = *(const unsigned int*)(vtb + (size_t)m * NFRAMES + kc + kk);
            v1.d[v] = *(const unsigned int*)(vtb + (size_t)(16 + m) * NFRAMES + kc + kk);
        }
        o0 = __builtin_amdgcn_wmma_f32_16x16x32_bf16(false, p.v, false, v0.v, (short)0, o0, false, false);
        o1 = __builtin_amdgcn_wmma_f32_16x16x32_bf16(false, p.v, false, v1.v, (short)0, o1, false, false);
    }
    unsigned short* ob = out16 + (size_t)(b * NFRAMES + q0) * 256 + h * HDIM;
#pragma unroll
    for (int r = 0; r < 8; ++r) {
        ob[(size_t)(r + 8 * half) * 256 + m]      = f2bf(o0[r]);
        ob[(size_t)(r + 8 * half) * 256 + 16 + m] = f2bf(o1[r]);
    }
}

// ---------------- control post-processing: tanh / exp ----------------
__global__ void k_postctrl(float* __restrict__ e) {
    int t = blockIdx.x, tid = threadIdx.x;
    for (int j = tid; j < 1024; j += 256) {
        float v = e[(size_t)t * 1024 + j];
        if (j < 256)      v = PI_F * tanhf(v);
        else if (j < 768) v = __expf(v);
        else              v = __expf(v) * (1.0f / 128.0f);
        e[(size_t)t * 1024 + j] = v;
    }
}

// ---------------- f0 upsample + wrapped cumsum phase + combtooth ----------------
__global__ void __launch_bounds__(512)
k_rot(const float* __restrict__ f0f, float* __restrict__ comb, float* __restrict__ phase) {
    __shared__ float s[512];
    __shared__ float carry_s;
    int b = blockIdx.x, tid = threadIdx.x;
    if (tid == 0) carry_s = 0.f;
    __syncthreads();
    for (int f = 0; f < NFRAMES; ++f) {
        float f00 = f0f[b * NFRAMES + f];
        float f01 = f0f[b * NFRAMES + ((f + 1 < NFRAMES) ? f + 1 : NFRAMES - 1)];
        float w   = (float)tid * (1.0f / 512.0f);
        float f0t = f00 + (f01 - f00) * w;
        float val = f0t * (1.0f / SRATE);
        s[tid] = val; __syncthreads();
        for (int o = 1; o < 512; o <<= 1) {             // inclusive Hillis-Steele scan
            float t = (tid >= o) ? s[tid - o] : 0.f;
            __syncthreads();
            s[tid] += t;
            __syncthreads();
        }
        float carry = carry_s;
        float rot = carry + s[tid];
        float wr = rot - rintf(rot);                    // wrap (round-half-even like jnp.round)
        if (tid == 0) phase[b * NFRAMES + f] = 2.0f * PI_F * wr;
        float x = SRATE * wr / (f0t + 0.001f);
        float c = (x == 0.f) ? 1.0f : sinf(PI_F * x) / (PI_F * x);   // np.sinc
        comb[(size_t)b * TLEN + (size_t)f * 512 + tid] = c;
        __syncthreads();
        if (tid == 511) { float nc = carry + s[511]; carry_s = nc - rintf(nc); }
        __syncthreads();
    }
}

// ---------------- in-LDS 2048-pt FFT: DIF fwd / DIT inv (no bit-reversal) ----------------
__device__ __forceinline__ void fft2048_dif(float* re, float* im, float sgn) {
    for (int sh = 10; sh >= 0; --sh) {
        int len = 1 << sh;
        __syncthreads();
        for (int idx = threadIdx.x; idx < 1024; idx += 256) {
            int j = idx & (len - 1);
            int i0 = ((idx >> sh) << (sh + 1)) + j, i1 = i0 + len;
            float ang = sgn * PI_F * (float)j / (float)len;
            float sv, cv; __sincosf(ang, &sv, &cv);
            float ar = re[i0], ai = im[i0], br = re[i1], bi = im[i1];
            re[i0] = ar + br; im[i0] = ai + bi;
            float dr = ar - br, di = ai - bi;
            re[i1] = dr * cv - di * sv;
            im[i1] = dr * sv + di * cv;
        }
    }
    __syncthreads();
}
__device__ __forceinline__ void fft2048_dit(float* re, float* im, float sgn) {
    for (int sh = 0; sh <= 10; ++sh) {
        int len = 1 << sh;
        __syncthreads();
        for (int idx = threadIdx.x; idx < 1024; idx += 256) {
            int j = idx & (len - 1);
            int i0 = ((idx >> sh) << (sh + 1)) + j, i1 = i0 + len;
            float ang = sgn * PI_F * (float)j / (float)len;
            float sv, cv; __sincosf(ang, &sv, &cv);
            float br = re[i1] * cv - im[i1] * sv;
            float bi = re[i1] * sv + im[i1] * cv;
            re[i1] = re[i0] - br; im[i1] = im[i0] - bi;
            re[i0] += br;         im[i0] += bi;
        }
    }
    __syncthreads();
}

// ---------------- per-frame time-varying FIR filter via spectral multiply ----------------
// MODE 0: allpass exp(i*cumsum(gd)) no window; MODE 1: exp(hm), variable raised-cosine;
// MODE 2: exp(nm)/128, fixed hann. yf gets the full 2048-sample frame convolution.
template <int NMAG, int MODE>
__global__ void __launch_bounds__(256)
k_filter(const float* __restrict__ audio, const float* __restrict__ ctrl,
         const float* __restrict__ f0f, float* __restrict__ yf) {
    constexpr int N = 2 * (NMAG - 1);          // IR length (510 or 1022)
    constexpr int NFFT = 2048;
    __shared__ float cosT[1022], sinT[1022];
    __shared__ float cre[512], cim[512];
    __shared__ float Ar[NFFT], Ai[NFFT], Br[NFFT], Bi[NFFT];
    int blk = blockIdx.x;
    int b = blk / NFRAMES, f = blk - b * NFRAMES;
    int tid = threadIdx.x;
    size_t trow = (size_t)blk * 1024;

    if (MODE == 0) {   // cumsum of group delay -> exp(i*phi)
        Ar[tid] = ctrl[trow + tid];
        __syncthreads();
        for (int o = 1; o < 256; o <<= 1) {
            float t = (tid >= o) ? Ar[tid - o] : 0.f;
            __syncthreads();
            Ar[tid] += t;
            __syncthreads();
        }
        float phi = Ar[tid];
        cre[tid] = cosf(phi); cim[tid] = sinf(phi);
    } else {
        int off = (MODE == 1) ? 256 : 768;
        for (int k = tid; k < NMAG; k += 256) { cre[k] = ctrl[trow + off + k]; cim[k] = 0.f; }
    }
    for (int i = tid; i < N; i += 256) {
        float th = 2.0f * PI_F * (float)i / (float)N;
        float sv, cv; __sincosf(th, &sv, &cv);
        cosT[i] = cv; sinT[i] = sv;
    }
    __syncthreads();

    float hw = 0.f;
    if (MODE == 1) hw = 1.5f * SRATE / (f0f[blk] + 0.001f);
    for (int i = tid; i < NFFT; i += 256) {
        float av = 0.f;
        if (i < 512) {
            float x = audio[(size_t)b * TLEN + (size_t)f * 512 + i];
            if (MODE == 2) x = 2.0f * x - 1.0f;
            av = x;
        }
        float bv = 0.f;
        if (i < N) {
            int tr = i + N / 2; if (tr >= N) tr -= N;   // roll(ir, N/2)
            // inverse real DFT (direct) with LDS twiddle tables
            float s = cre[0] + ((tr & 1) ? -cre[NMAG - 1] : cre[NMAG - 1]);
            int idx = tr;
            for (int k = 1; k <= NMAG - 2; ++k) {
                s += 2.0f * (cre[k] * cosT[idx] - cim[k] * sinT[idx]);
                idx += tr; if (idx >= N) idx -= N;
            }
            float ir = s * (1.0f / (float)N);
            float w = 1.0f;
            if (MODE == 1) {
                float tt = (float)(i - N / 2);
                float r = tt / hw; r = fminf(1.f, fmaxf(-1.f, r));
                w = 0.5f + 0.5f * cosf(PI_F * r);
            } else if (MODE == 2) {
                w = 0.5f - 0.5f * cosT[i];              // hann over ir_size
            }
            bv = ir * w;
        }
        Ar[i] = av; Ai[i] = 0.f; Br[i] = bv; Bi[i] = 0.f;
    }
    __syncthreads();
    fft2048_dif(Ar, Ai, -1.0f);
    fft2048_dif(Br, Bi, -1.0f);
    for (int i = tid; i < NFFT; i += 256) {             // multiply in bit-reversed domain
        float xr = Ar[i], xi = Ai[i], yr = Br[i], yi = Bi[i];
        Ar[i] = xr * yr - xi * yi; Ai[i] = xr * yi + xi * yr;
    }
    __syncthreads();
    fft2048_dit(Ar, Ai, 1.0f);
    const float sc = 1.0f / (float)NFFT;
    for (int i = tid; i < NFFT; i += 256)
        yf[(size_t)blk * NFFT + i] = Ar[i] * sc;
}

// deterministic overlap-add: gather <=4 overlapping frame convolutions, shift by delay
__global__ void k_overlap_gather(const float* __restrict__ yf, float* __restrict__ out, int delay) {
    int i = blockIdx.x * 256 + threadIdx.x;
    if (i >= BATCH * TLEN) return;
    int b = i / TLEN, t = i - b * TLEN;
    int s = t + delay;
    int f_hi = s >> 9;
    float acc = 0.f;
#pragma unroll
    for (int d = 0; d < 4; ++d) {
        int f = f_hi - d;
        int off = s - f * 512;
        if (f >= 0 && f < NFRAMES && off < 2048)
            acc += yf[((size_t)(b * NFRAMES + f)) * 2048 + off];
    }
    out[i] = acc;
}

__global__ void k_add(const float* __restrict__ x, const float* __restrict__ y,
                      float* __restrict__ z, int n) {
    int i = blockIdx.x * 256 + threadIdx.x;
    if (i < n) z[i] = x[i] + y[i];
}

// ---------------- host-side orchestration ----------------
extern "C" void kernel_launch(void* const* d_in, const int* in_sizes, int n_in,
                              void* d_out, int out_size, void* d_ws, size_t ws_size,
                              hipStream_t stream) {
    const float* units = (const float*)d_in[0];
    const float* f0f   = (const float*)d_in[1];
    const float* vol   = (const float*)d_in[2];
    const float* noise = (const float*)d_in[3];
    // d_in[4] spk_id unused (n_spk==1)
    const float* conv1_w = (const float*)d_in[5];
    const float* conv1_b = (const float*)d_in[6];
    const float* gn_g = (const float*)d_in[7];
    const float* gn_b = (const float*)d_in[8];
    const float* conv2_w = (const float*)d_in[9];
    const float* conv2_b = (const float*)d_in[10];
    const float* f0_w = (const float*)d_in[11];
    const float* f0_b = (const float*)d_in[12];
    const float* ph_w = (const float*)d_in[13];
    const float* ph_b = (const float*)d_in[14];
    const float* vw   = (const float*)d_in[15];
    const float* vb   = (const float*)d_in[16];
    const float* lnf_g = (const float*)d_in[53];
    const float* lnf_b = (const float*)d_in[54];
    const float* out_w = (const float*)d_in[55];
    const float* out_b = (const float*)d_in[56];

    char* wsp = (char*)d_ws;
    auto alloc = [&](size_t bytes) -> char* {
        char* p = wsp; wsp += (bytes + 255) & ~(size_t)255; return p;
    };
    float* comb  = (float*)alloc((size_t)BATCH * TLEN * 4);
    float* harm1 = (float*)alloc((size_t)BATCH * TLEN * 4);
    float* yf    = (float*)alloc((size_t)NTOK * 2048 * 4);
    float* xa    = (float*)alloc((size_t)NTOK * 256 * 4);
    float* xb    = (float*)alloc((size_t)NTOK * 256 * 4);
    unsigned short* xn16  = (unsigned short*)alloc((size_t)NTOK * 256 * 2);
    unsigned short* tok16 = (unsigned short*)alloc((size_t)NTOK * 768 * 2);
    unsigned short* vt16  = (unsigned short*)alloc((size_t)NTOK * 256 * 2);
    float* ffh   = (float*)alloc((size_t)NTOK * 1024 * 4);
    unsigned short* ffh16 = (unsigned short*)alloc((size_t)NTOK * 1024 * 2);
    unsigned int* Wc1 = (unsigned int*)alloc(384 * 256 * 4);
    unsigned int* Wc2 = (unsigned int*)alloc(384 * 256 * 4);
    unsigned int *Wqkv[3], *Wo[3], *W1[3], *W2[3];
    for (int L = 0; L < 3; ++L) {
        Wqkv[L] = (unsigned int*)alloc(128 * 768 * 4);
        Wo[L]   = (unsigned int*)alloc(128 * 256 * 4);
        W1[L]   = (unsigned int*)alloc(128 * 1024 * 4);
        W2[L]   = (unsigned int*)alloc(512 * 256 * 4);
    }
    unsigned int* Wout = (unsigned int*)alloc(128 * 1024 * 4);

    float* o_sig   = (float*)d_out;
    float* o_ph    = o_sig + (size_t)BATCH * TLEN;
    float* o_harm  = o_ph + (size_t)BATCH * NFRAMES;
    float* o_noise = o_harm + (size_t)BATCH * TLEN;

    auto pack = [&](const float* src, unsigned int* dst, int K, int N) {
        int n = (K / 2) * N;
        k_pack_pairs<<<(n + 255) / 256, 256, 0, stream>>>(src, dst, K, N);
    };
    // ----- weight packing (re-done each call; deterministic) -----
    k_pack_conv_pairs<<<(384 * 256 + 255) / 256, 256, 0, stream>>>(conv1_w, Wc1);
    k_pack_conv_pairs<<<(384 * 256 + 255) / 256, 256, 0, stream>>>(conv2_w, Wc2);
    for (int L = 0; L < 3; ++L) {
        int base = 17 + 12 * L;
        pack((const float*)d_in[base + 2],  Wqkv[L], 256, 768);
        pack((const float*)d_in[base + 4],  Wo[L],   256, 256);
        pack((const float*)d_in[base + 8],  W1[L],   256, 1024);
        pack((const float*)d_in[base + 10], W2[L],   1024, 256);
    }
    pack(out_w, Wout, 256, 1024);

    // ----- phase / combtooth -----
    k_rot<<<BATCH, 512, 0, stream>>>(f0f, comb, o_ph);

    // ----- Unit2Control -----
    k_im2col<<<NTOK, 256, 0, stream>>>(units, tok16);
    launch_gemm<256, 768, 0>(tok16, Wc1, conv1_b, nullptr, xb, nullptr, NTOK, stream);
    k_groupnorm_leaky<<<BATCH * 4, 256, 0, stream>>>(xb, gn_g, gn_b, xa);
    k_im2col<<<NTOK, 256, 0, stream>>>(xa, tok16);
    launch_gemm<256, 768, 0>(tok16, Wc2, conv2_b, nullptr, xa, nullptr, NTOK, stream);
    k_cond_add<<<NTOK, 256, 0, stream>>>(xa, f0f, o_ph, vol, f0_w, f0_b, ph_w, ph_b, vw, vb);

    for (int L = 0; L < 3; ++L) {
        int base = 17 + 12 * L;
        const float* ln1g = (const float*)d_in[base + 0];
        const float* ln1b = (const float*)d_in[base + 1];
        const float* bqkv = (const float*)d_in[base + 3];
        const float* bo   = (const float*)d_in[base + 5];
        const float* ln2g = (const float*)d_in[base + 6];
        const float* ln2b = (const float*)d_in[base + 7];
        const float* b1   = (const float*)d_in[base + 9];
        const float* b2   = (const float*)d_in[base + 11];
        k_layernorm<<<NTOK, 256, 0, stream>>>(xa, ln1g, ln1b, xn16);
        launch_gemm<768, 256, 0>(xn16, Wqkv[L], bqkv, nullptr, nullptr, tok16, NTOK, stream);
        k_vt<<<(NTOK * 256 + 255) / 256, 256, 0, stream>>>(tok16, vt16);
        k_attn<<<BATCH * HEADS * (NFRAMES / 16), 32, 0, stream>>>(tok16, vt16, xn16);
        launch_gemm<256, 256, 0>(xn16, Wo[L], bo, xa, xa, nullptr, NTOK, stream);
        k_layernorm<<<NTOK, 256, 0, stream>>>(xa, ln2g, ln2b, xn16);
        launch_gemm<1024, 256, 2>(xn16, W1[L], b1, nullptr, nullptr, ffh16, NTOK, stream);
        launch_gemm<256, 1024, 0>(ffh16, W2[L], b2, xa, xa, nullptr, NTOK, stream);
    }
    k_layernorm<<<NTOK, 256, 0, stream>>>(xa, lnf_g, lnf_b, xn16);
    launch_gemm<1024, 256, 0>(xn16, Wout, out_b, nullptr, ffh, nullptr, NTOK, stream);
    k_postctrl<<<NTOK, 256, 0, stream>>>(ffh);

    // ----- filtering pipeline -----
    int ng = (BATCH * TLEN + 255) / 256;
    k_filter<256, 0><<<NTOK, 256, 0, stream>>>(comb, ffh, f0f, yf);     // allpass comb
    k_overlap_gather<<<ng, 256, 0, stream>>>(yf, harm1, 255);
    k_filter<512, 1><<<NTOK, 256, 0, stream>>>(harm1, ffh, f0f, yf);    // harmonic shaping
    k_overlap_gather<<<ng, 256, 0, stream>>>(yf, o_harm, 511);
    k_filter<256, 2><<<NTOK, 256, 0, stream>>>(noise, ffh, f0f, yf);    // noise shaping
    k_overlap_gather<<<ng, 256, 0, stream>>>(yf, o_noise, 255);
    k_add<<<ng, 256, 0, stream>>>(o_harm, o_noise, o_sig, BATCH * TLEN);
}
